// MultiQueryAttention_27539330302255
// MI455X (gfx1250) — matrix operand order repaired and checked by reading, not compiled, not run on hardware
//
#include <hip/hip_runtime.h>
#include <hip/hip_bf16.h>
#include <stdint.h>

// Problem constants (from reference)
#define B_   2
#define S_   2048
#define DIM_ 2048
#define H_   16
#define HD_  128
#define BS_  (B_*S_)

typedef __attribute__((ext_vector_type(16))) __bf16 v16bf;
typedef __attribute__((ext_vector_type(8)))  __bf16 v8bf;
typedef __attribute__((ext_vector_type(4)))  __bf16 v4bf;
typedef __attribute__((ext_vector_type(8)))  float  v8f;
typedef __attribute__((ext_vector_type(4)))  float  v4f;

__device__ __forceinline__ __bf16 f2bf(float f) { return static_cast<__bf16>(f); }
__device__ __forceinline__ void st_out(float* p, float v)  { *p = v; }
__device__ __forceinline__ void st_out(__bf16* p, float v) { *p = f2bf(v); }

// Low 32 bits of a generic pointer to __shared__ = LDS byte offset (aperture
// semantics) — what GLOBAL_LOAD_ASYNC_TO_LDS wants.
__device__ __forceinline__ unsigned lds_off(const void* p) {
  return (unsigned)(uintptr_t)p;
}
__device__ __forceinline__ void async_copy_b128(unsigned ldsaddr, const void* gaddr) {
  asm volatile("global_load_async_to_lds_b128 %0, %1, off"
               :: "v"(ldsaddr), "v"((unsigned long long)(uintptr_t)gaddr)
               : "memory");
}
__device__ __forceinline__ void wait_async0() {
  asm volatile("s_wait_asynccnt 0x0" ::: "memory");
}

// ---------------------------------------------------------------------------
// Generic GEMM: out[M,N] = A[M,K] * W[K,N] + bias[N]
// Block tile 128x128, K-step 32, 256 threads = 8 waves, double-buffered LDS.
// A staged (m,k) bf16 (async DMA when already bf16, else f32->bf16 via VGPR);
// W staged transposed (n,k). One barrier per K-step; next tile's global
// traffic is issued before the current tile's WMMAs (full-phase latency cover).
// ---------------------------------------------------------------------------
template<typename AT, typename OT>
__global__ __launch_bounds__(256)
void gemm_wmma_bf16(const AT* __restrict__ A, const float* __restrict__ W,
                    const float* __restrict__ bias, OT* __restrict__ out,
                    int M, int N, int K)
{
  __shared__ __align__(16) __bf16 a_lds[2][128][40];   // [buf][m][k]
  __shared__ __align__(16) __bf16 b_lds[2][128][40];   // [buf][n][k] transposed

  const int t    = threadIdx.x;
  const int wave = t >> 5;
  const int lane = t & 31;
  const int lh   = lane >> 4;
  const int ll   = lane & 15;

  const int m0 = blockIdx.y * 128;
  const int n0 = blockIdx.x * 128;

  constexpr bool ASYNC_A = (sizeof(AT) == 2);

  v8f acc[8];
  #pragma unroll
  for (int j = 0; j < 8; ++j)
    #pragma unroll
    for (int e = 0; e < 8; ++e) acc[j][e] = 0.0f;

  // ---- staging helpers ----
  auto issueA = [&](int k0, int buf) {          // bf16 source: async DMA
    #pragma unroll
    for (int i = 0; i < 2; ++i) {
      int chunk = i * 256 + t;                  // 512 chunks of 16B
      int r = chunk >> 2, c8 = (chunk & 3) * 8;
      async_copy_b128(lds_off(&a_lds[buf][r][c8]),
                      &A[(size_t)(m0 + r) * K + (k0 + c8)]);
    }
  };
  auto loadA = [&](int k0, v4f* areg) {         // f32 source: load to regs
    #pragma unroll
    for (int i = 0; i < 4; ++i) {
      int chunk = i * 256 + t;                  // 1024 chunks of 4 floats
      int r = chunk >> 3, c4 = (chunk & 7) * 4;
      areg[i] = *(const v4f*)&A[(size_t)(m0 + r) * K + (k0 + c4)];
    }
  };
  auto storeA = [&](int buf, const v4f* areg) { // cvt + LDS store
    #pragma unroll
    for (int i = 0; i < 4; ++i) {
      int chunk = i * 256 + t;
      int r = chunk >> 3, c4 = (chunk & 7) * 4;
      *(v4bf*)&a_lds[buf][r][c4] = __builtin_convertvector(areg[i], v4bf);
    }
  };
  auto loadW = [&](int k0, v4f* wreg) {
    #pragma unroll
    for (int i = 0; i < 4; ++i) {
      int chunk = i * 256 + t;
      int r = chunk >> 5, c4 = (chunk & 31) * 4;
      wreg[i] = *(const v4f*)&W[(size_t)(k0 + r) * N + (n0 + c4)];
    }
  };
  auto storeW = [&](int buf, const v4f* wreg) { // cvt + transposed LDS store
    #pragma unroll
    for (int i = 0; i < 4; ++i) {
      int chunk = i * 256 + t;
      int r = chunk >> 5, c4 = (chunk & 31) * 4;
      v4bf wb = __builtin_convertvector(wreg[i], v4bf);
      #pragma unroll
      for (int e = 0; e < 4; ++e) b_lds[buf][c4 + e][r] = wb[e];
    }
  };

  v4f aTmp[4], wTmp[4];
  const int NT = K / 32;

  // prologue: stage tile 0 into buffer 0
  if constexpr (ASYNC_A) issueA(0, 0);
  else { loadA(0, aTmp); storeA(0, aTmp); }
  loadW(0, wTmp); storeW(0, wTmp);

  for (int it = 0; it < NT; ++it) {
    const int cur = it & 1, nxt = cur ^ 1;
    if constexpr (ASYNC_A) wait_async0();       // tile it's DMA complete
    __syncthreads();                            // visible to all waves
    const bool more = (it + 1 < NT);
    if (more) {
      const int k1 = (it + 1) * 32;
      if constexpr (ASYNC_A) issueA(k1, nxt);   // overlap with compute below
      else loadA(k1, aTmp);
      loadW(k1, wTmp);
      if (it + 2 < NT) {
        __builtin_prefetch(&A[(size_t)(m0 + (t >> 3)) * K + (it + 2) * 32], 0, 1);
        __builtin_prefetch(&W[(size_t)((it + 2) * 32 + (t >> 5)) * N + n0 + (t & 31) * 4], 0, 1);
      }
    }

    // ---- compute on buffer cur ----
    v16bf af;
    {
      int arow = wave * 16 + ll;
      v8bf lo = *(const v8bf*)&a_lds[cur][arow][lh * 8];
      v8bf hi = *(const v8bf*)&a_lds[cur][arow][16 + lh * 8];
      #pragma unroll
      for (int e = 0; e < 8; ++e) { af[e] = lo[e]; af[8 + e] = hi[e]; }
    }
    #pragma unroll
    for (int j = 0; j < 8; ++j) {
      int bcol = j * 16 + ll;
      v8bf lo = *(const v8bf*)&b_lds[cur][bcol][lh * 8];
      v8bf hi = *(const v8bf*)&b_lds[cur][bcol][16 + lh * 8];
      v16bf bfr;
      #pragma unroll
      for (int e = 0; e < 8; ++e) { bfr[e] = lo[e]; bfr[8 + e] = hi[e]; }
      acc[j] = __builtin_amdgcn_wmma_f32_16x16x32_bf16(
          false, af, false, bfr, (short)0, acc[j], false, false);
    }

    if (more) {                                 // fill the other buffer
      if constexpr (!ASYNC_A) storeA(nxt, aTmp);
      storeW(nxt, wTmp);
    }
  }

  // epilogue: C layout — lane half selects rows r / r+8
  const int orow = m0 + wave * 16 + lh * 8;
  #pragma unroll
  for (int j = 0; j < 8; ++j) {
    int ocol = n0 + j * 16 + ll;
    float bv = bias[ocol];
    #pragma unroll
    for (int r = 0; r < 8; ++r)
      st_out(&out[(size_t)(orow + r) * N + ocol], acc[j][r] + bv);
  }
}

// ---------------------------------------------------------------------------
// Flash-attention MQA kernel. One block per (q-tile of 128, head, batch).
// 8 waves; each wave owns 16 query rows. Double-buffered K/V tiles of 32 keys:
// K async-DMA'd to LDS, V transposed through VGPR temps, one barrier per tile.
// ---------------------------------------------------------------------------
__global__ __launch_bounds__(256)
void mqa_flash_wmma(const __bf16* __restrict__ Q,    // [B*S, H*HD]
                    const __bf16* __restrict__ Km,   // [B*S, HD]
                    const __bf16* __restrict__ Vm,   // [B*S, HD]
                    const float*  __restrict__ mask, // [B,1,S,S]
                    __bf16* __restrict__ O)          // [B*S, H*HD]
{
  __shared__ __align__(16) __bf16 k_lds[2][32][136];   // [buf][key][d]
  __shared__ __align__(16) __bf16 vt_lds[2][128][40];  // [buf][d][key]
  __shared__ __align__(16) __bf16 p_lds[8][16][40];    // per-wave P tile

  const int t    = threadIdx.x;
  const int wave = t >> 5;
  const int lane = t & 31;
  const int lh   = lane >> 4;
  const int ll   = lane & 15;

  const int qt = blockIdx.x * 128;
  const int h  = blockIdx.y;
  const int b  = blockIdx.z;

  const float scale = 0.08838834764831845f; // 1/sqrt(128)

  auto issueK = [&](int k0, int buf) {
    #pragma unroll
    for (int i = 0; i < 2; ++i) {
      int chunk = i * 256 + t;
      int r = chunk >> 4, c8 = (chunk & 15) * 8;
      async_copy_b128(lds_off(&k_lds[buf][r][c8]),
                      &Km[(size_t)(b * S_ + k0 + r) * HD_ + c8]);
    }
  };
  auto loadV = [&](int k0, v8bf* vreg) {
    #pragma unroll
    for (int i = 0; i < 2; ++i) {
      int chunk = i * 256 + t;
      int r = chunk >> 4, c8 = (chunk & 15) * 8;
      vreg[i] = *(const v8bf*)&Vm[(size_t)(b * S_ + k0 + r) * HD_ + c8];
    }
  };
  auto storeV = [&](int buf, const v8bf* vreg) {
    #pragma unroll
    for (int i = 0; i < 2; ++i) {
      int chunk = i * 256 + t;
      int r = chunk >> 4, c8 = (chunk & 15) * 8;
      #pragma unroll
      for (int e = 0; e < 8; ++e) vt_lds[buf][c8 + e][r] = vreg[i][e];
    }
  };

  // Preload this wave's Q fragments: 4 d-chunks of 32 (reused S/32 times)
  v16bf qf[4];
  {
    int qrow = qt + wave * 16 + ll;
    const __bf16* qp = Q + (size_t)(b * S_ + qrow) * (H_ * HD_) + h * HD_;
    #pragma unroll
    for (int dc = 0; dc < 4; ++dc) {
      v8bf lo = *(const v8bf*)(qp + dc * 32 + lh * 8);
      v8bf hi = *(const v8bf*)(qp + dc * 32 + 16 + lh * 8);
      #pragma unroll
      for (int e = 0; e < 8; ++e) { qf[dc][e] = lo[e]; qf[dc][8 + e] = hi[e]; }
    }
  }

  v8f acc[8];
  #pragma unroll
  for (int j = 0; j < 8; ++j)
    #pragma unroll
    for (int e = 0; e < 8; ++e) acc[j][e] = 0.0f;

  float m_r[8], l_r[8];
  #pragma unroll
  for (int r = 0; r < 8; ++r) { m_r[r] = -3.0e38f; l_r[r] = 0.0f; }

  const size_t mrow0 = ((size_t)b * S_ + (size_t)(qt + wave * 16 + lh * 8)) * S_;

  v8bf vTmp[2];
  const int NT = S_ / 32;

  // prologue: stage tile 0
  issueK(0, 0);
  loadV(0, vTmp); storeV(0, vTmp);

  for (int it = 0; it < NT; ++it) {
    const int cur = it & 1, nxt = cur ^ 1;
    const int k0 = it * 32;
    wait_async0();
    __syncthreads();
    const bool more = (it + 1 < NT);
    if (more) {
      issueK(k0 + 32, nxt);
      loadV(k0 + 32, vTmp);
      if (it + 2 < NT) {
        __builtin_prefetch(&Km[(size_t)(b * S_ + k0 + 64 + (t >> 4)) * HD_ + (t & 15) * 8], 0, 1);
        __builtin_prefetch(&Vm[(size_t)(b * S_ + k0 + 64 + (t >> 4)) * HD_ + (t & 15) * 8], 0, 1);
      }
    }

    // scores S = Q * K^T : [16 q] x [32 keys], Kdim = 128 (4 wmma K-steps)
    v8f s[2];
    #pragma unroll
    for (int c = 0; c < 2; ++c)
      #pragma unroll
      for (int e = 0; e < 8; ++e) s[c][e] = 0.0f;
    #pragma unroll
    for (int c = 0; c < 2; ++c) {
      #pragma unroll
      for (int dc = 0; dc < 4; ++dc) {
        v8bf lo = *(const v8bf*)&k_lds[cur][c * 16 + ll][dc * 32 + lh * 8];
        v8bf hi = *(const v8bf*)&k_lds[cur][c * 16 + ll][dc * 32 + 16 + lh * 8];
        v16bf kb;
        #pragma unroll
        for (int e = 0; e < 8; ++e) { kb[e] = lo[e]; kb[8 + e] = hi[e]; }
        s[c] = __builtin_amdgcn_wmma_f32_16x16x32_bf16(
            false, qf[dc], false, kb, (short)0, s[c], false, false);
      }
    }

    // online softmax (rows live in 16-lane halves of the C layout)
    float corr[8];
    #pragma unroll
    for (int r = 0; r < 8; ++r) {
      float s0 = s[0][r] * scale + mask[mrow0 + (size_t)r * S_ + k0 + ll];
      float s1 = s[1][r] * scale + mask[mrow0 + (size_t)r * S_ + k0 + 16 + ll];
      float mx = fmaxf(s0, s1);
      mx = fmaxf(mx, __shfl_xor(mx, 1));
      mx = fmaxf(mx, __shfl_xor(mx, 2));
      mx = fmaxf(mx, __shfl_xor(mx, 4));
      mx = fmaxf(mx, __shfl_xor(mx, 8));
      float mn = fmaxf(m_r[r], mx);
      float p0 = __expf(s0 - mn);
      float p1 = __expf(s1 - mn);
      float sum = p0 + p1;
      sum += __shfl_xor(sum, 1);
      sum += __shfl_xor(sum, 2);
      sum += __shfl_xor(sum, 4);
      sum += __shfl_xor(sum, 8);
      float cr = __expf(m_r[r] - mn);
      l_r[r] = l_r[r] * cr + sum;
      m_r[r] = mn;
      corr[r] = cr;
      p_lds[wave][lh * 8 + r][ll]      = f2bf(p0);
      p_lds[wave][lh * 8 + r][16 + ll] = f2bf(p1);
    }
    #pragma unroll
    for (int j = 0; j < 8; ++j)
      #pragma unroll
      for (int r = 0; r < 8; ++r) acc[j][r] *= corr[r];

    asm volatile("s_wait_dscnt 0x0" ::: "memory");

    // P as A fragment (16 q x 32 keys)
    v16bf pf;
    {
      v8bf lo = *(const v8bf*)&p_lds[wave][ll][lh * 8];
      v8bf hi = *(const v8bf*)&p_lds[wave][ll][16 + lh * 8];
      #pragma unroll
      for (int e = 0; e < 8; ++e) { pf[e] = lo[e]; pf[8 + e] = hi[e]; }
    }
    // acc += P * V : N = 128 d (8 subtiles), Kdim = 32 keys
    #pragma unroll
    for (int j = 0; j < 8; ++j) {
      v8bf lo = *(const v8bf*)&vt_lds[cur][j * 16 + ll][lh * 8];
      v8bf hi = *(const v8bf*)&vt_lds[cur][j * 16 + ll][16 + lh * 8];
      v16bf vfr;
      #pragma unroll
      for (int e = 0; e < 8; ++e) { vfr[e] = lo[e]; vfr[8 + e] = hi[e]; }
      acc[j] = __builtin_amdgcn_wmma_f32_16x16x32_bf16(
          false, pf, false, vfr, (short)0, acc[j], false, false);
    }

    if (more) storeV(nxt, vTmp);   // fill other buffer after reads of it ended
  }

  // epilogue: divide by l, store bf16 into attn buffer [b, s, h*d]
  const int qrow = qt + wave * 16 + lh * 8;
  #pragma unroll
  for (int j = 0; j < 8; ++j) {
    #pragma unroll
    for (int r = 0; r < 8; ++r) {
      float o = acc[j][r] / l_r[r];
      O[(size_t)(b * S_ + qrow + r) * (H_ * HD_) + h * HD_ + j * 16 + ll] = f2bf(o);
    }
  }
}

// ---------------------------------------------------------------------------
// Host launcher
// Inputs: hidden(0) mask(1) wq(2) bq(3) wk(4) bk(5) wv(6) bv(7) wo(8) bo(9)
// ---------------------------------------------------------------------------
extern "C" void kernel_launch(void* const* d_in, const int* in_sizes, int n_in,
                              void* d_out, int out_size, void* d_ws, size_t ws_size,
                              hipStream_t stream)
{
  const float* hidden = (const float*)d_in[0];
  const float* amask  = (const float*)d_in[1];
  const float* wq = (const float*)d_in[2];
  const float* bq = (const float*)d_in[3];
  const float* wk = (const float*)d_in[4];
  const float* bk = (const float*)d_in[5];
  const float* wv = (const float*)d_in[6];
  const float* bv = (const float*)d_in[7];
  const float* wo = (const float*)d_in[8];
  const float* bo = (const float*)d_in[9];
  float* out = (float*)d_out;

  char* ws = (char*)d_ws;
  size_t off = 0;
  __bf16* qb  = (__bf16*)(ws + off); off += (size_t)BS_ * DIM_ * 2;   // Q  bf16
  __bf16* kb  = (__bf16*)(ws + off); off += (size_t)BS_ * HD_  * 2;   // K  bf16
  __bf16* vb  = (__bf16*)(ws + off); off += (size_t)BS_ * HD_  * 2;   // V  bf16
  __bf16* atb = (__bf16*)(ws + off); off += (size_t)BS_ * DIM_ * 2;   // attn bf16

  dim3 blk(256);

  // Q/K/V projections (f32 in -> bf16 out)
  gemm_wmma_bf16<float, __bf16><<<dim3(DIM_ / 128, BS_ / 128), blk, 0, stream>>>(
      hidden, wq, bq, qb, BS_, DIM_, DIM_);
  gemm_wmma_bf16<float, __bf16><<<dim3(HD_ / 128, BS_ / 128), blk, 0, stream>>>(
      hidden, wk, bk, kb, BS_, HD_, DIM_);
  gemm_wmma_bf16<float, __bf16><<<dim3(HD_ / 128, BS_ / 128), blk, 0, stream>>>(
      hidden, wv, bv, vb, BS_, HD_, DIM_);

  // flash attention
  mqa_flash_wmma<<<dim3(S_ / 128, H_, B_), blk, 0, stream>>>(qb, kb, vb, amask, atb);

  // output projection (bf16 in -> f32 out)
  gemm_wmma_bf16<__bf16, float><<<dim3(DIM_ / 128, BS_ / 128), blk, 0, stream>>>(
      atb, wo, bo, out, BS_, DIM_, DIM_);
}
